// MultiHeadSelfAttention_84430467105211
// MI455X (gfx1250) — matrix operand order repaired and checked by reading, not compiled, not run on hardware
//
#include <hip/hip_runtime.h>

#define B_ 2
#define T_ 4096
#define C_ 768
#define H_ 12
#define D_ 64

typedef __attribute__((ext_vector_type(16))) __bf16 v16bf;
typedef __attribute__((ext_vector_type(8)))  float  v8f;

union FragCvt { uint4 q[2]; v16bf v; };

// native fp32 -> bf16 (RNE) — single hardware cvt on gfx1250
__device__ __forceinline__ unsigned short f2bf(float f) {
  __bf16 h = (__bf16)f;
  return __builtin_bit_cast(unsigned short, h);
}

// A-operand fragment (16x32 bf16): lane holds row (lr), half selects K-subset
// {0-7,16-23} vs {8-15,24-31}: two 16B LDS loads at +0 and +32 bytes.
__device__ __forceinline__ v16bf lds_frag_a(const unsigned short* p) {
  FragCvt f;
  f.q[0] = *(const uint4*)(p);
  f.q[1] = *(const uint4*)(p + 16);
  return f.v;
}
// B-operand fragment (32x16 bf16): lane holds column (lr), half selects K
// {0-7,16-23} vs {8-15,24-31}: two 16B LDS loads at +0 and +32 bytes.
__device__ __forceinline__ v16bf lds_frag_b(const unsigned short* p) {
  FragCvt f;
  f.q[0] = *(const uint4*)(p);
  f.q[1] = *(const uint4*)(p + 16);
  return f.v;
}

#define WMMA_BF16(a, b, c) \
  __builtin_amdgcn_wmma_f32_16x16x32_bf16(false, (a), false, (b), (short)0, (c), false, false)

// gfx1250 async global->LDS copy (ASYNCcnt path). VDST operand = per-lane LDS
// byte offset (low 32 bits of the generic LDS pointer), addr = 64-bit global.
__device__ __forceinline__ void async_ld_b128(void* lds, const void* g) {
  asm volatile("global_load_async_to_lds_b128 %0, %1, off"
               :: "v"((unsigned)(size_t)lds), "v"(g) : "memory");
}
__device__ __forceinline__ void wait_async0() {
  asm volatile("s_wait_asynccnt 0x0" ::: "memory");
}

// ---------------------------------------------------------------------------
// Kernel 1: qkv = x @ w_qkv + b_qkv, scattered to Q/K/V bf16 [B,H,T,D]
// ---------------------------------------------------------------------------
__global__ __launch_bounds__(256) void qkv_gemm_kernel(
    const float* __restrict__ x, const float* __restrict__ w,
    const float* __restrict__ bias,
    unsigned short* __restrict__ qb, unsigned short* __restrict__ kb,
    unsigned short* __restrict__ vb) {
  __shared__ alignas(16) unsigned short As[128 * 40];  // [row][k], k-stride 40
  __shared__ alignas(16) unsigned short Bs[128 * 40];  // [n][k]
  const int n0 = blockIdx.x * 128;
  const int m0 = blockIdx.y * 128;
  const int tid = threadIdx.x;
  const int wave = tid >> 5, lane = tid & 31;
  const int half = lane >> 4, lr = lane & 15;
  const int wr = wave & 3, wc = wave >> 2;  // 4 row-groups x 2 col-groups

  v8f acc[2][4] = {};

  for (int k0 = 0; k0 < C_; k0 += 32) {
    __syncthreads();
    {  // stage A tile: x[m0..m0+127][k0..k0+31], fp32 -> bf16
      const int tx = tid & 7, ty = tid >> 3;
#pragma unroll
      for (int i = 0; i < 4; ++i) {
        const int row = ty + i * 32;
        float4 v = *(const float4*)(x + (size_t)(m0 + row) * C_ + k0 + tx * 4);
        unsigned short* d = As + row * 40 + tx * 4;
        d[0] = f2bf(v.x); d[1] = f2bf(v.y); d[2] = f2bf(v.z); d[3] = f2bf(v.w);
      }
    }
    {  // stage B tile transposed: w[k0..k0+31][n0..n0+127] -> Bs[n][k]
      const int nx = tid & 31, ky = tid >> 5;
#pragma unroll
      for (int i = 0; i < 4; ++i) {
        const int k = ky + i * 8;
        float4 v = *(const float4*)(w + (size_t)(k0 + k) * (3 * C_) + n0 + nx * 4);
        Bs[(nx * 4 + 0) * 40 + k] = f2bf(v.x);
        Bs[(nx * 4 + 1) * 40 + k] = f2bf(v.y);
        Bs[(nx * 4 + 2) * 40 + k] = f2bf(v.z);
        Bs[(nx * 4 + 3) * 40 + k] = f2bf(v.w);
      }
    }
    __syncthreads();
    v16bf af[2], bf[4];
#pragma unroll
    for (int mt = 0; mt < 2; ++mt)
      af[mt] = lds_frag_a(As + (wr * 32 + mt * 16 + lr) * 40 + half * 8);
#pragma unroll
    for (int nt = 0; nt < 4; ++nt)
      bf[nt] = lds_frag_b(Bs + (wc * 64 + nt * 16 + lr) * 40 + half * 8);
#pragma unroll
    for (int mt = 0; mt < 2; ++mt)
#pragma unroll
      for (int nt = 0; nt < 4; ++nt)
        acc[mt][nt] = WMMA_BF16(af[mt], bf[nt], acc[mt][nt]);
  }

  // epilogue: which/head/batch are uniform per block/wave -> scalar math only
  const int which = n0 / C_;  // 128-wide tile fully inside Q, K, or V
  unsigned short* __restrict__ dst = (which == 0) ? qb : ((which == 1) ? kb : vb);
  const int cbase = (n0 % C_) + wc * 64;  // multiple of 64 -> single head
  const int h = cbase >> 6;
  const int bb = m0 >> 12;                // 128-row tile inside one batch
  const int trow = (m0 & (T_ - 1)) + wr * 32;
  const size_t base = (((size_t)(bb * H_ + h)) * T_ + trow) * D_;
#pragma unroll
  for (int nt = 0; nt < 4; ++nt) {
    const int d = nt * 16 + lr;
    const float bval = bias[n0 + wc * 64 + nt * 16 + lr];
#pragma unroll
    for (int mt = 0; mt < 2; ++mt)
#pragma unroll
      for (int r = 0; r < 8; ++r) {
        const int rr = mt * 16 + r + half * 8;  // row offset within wave slice
        dst[base + (size_t)rr * D_ + d] = f2bf(acc[mt][nt][r] + bval);
      }
  }
}

// ---------------------------------------------------------------------------
// Kernel 2: causal flash-attention; block = (b,h, 128-query tile), 8 waves,
// each wave owns 16 query rows (softmax rows stay within one wave-half).
// Q/K tiles arrive via gfx1250 async global->LDS; V is transposed manually.
// ---------------------------------------------------------------------------
__global__ __launch_bounds__(256) void attn_kernel(
    const unsigned short* __restrict__ qbuf,
    const unsigned short* __restrict__ kbuf,
    const unsigned short* __restrict__ vbuf,
    unsigned short* __restrict__ attn) {
  __shared__ alignas(16) unsigned short Qs[128 * 72];      // [q][d]
  __shared__ alignas(16) unsigned short Ks[128 * 72];      // [key][d]
  __shared__ alignas(16) unsigned short Vs[64 * 136];      // [d][key] (transposed)
  __shared__ alignas(16) unsigned short Ps[8 * 16 * 136];  // per-wave P slab

  const int qt = blockIdx.x & 31;  // T/128 = 32 query tiles
  const int bh = blockIdx.x >> 5;
  const int b = bh / H_, h = bh % H_;
  const int q0 = qt * 128;
  const int tid = threadIdx.x;
  const int wave = tid >> 5, lane = tid & 31;
  const int half = lane >> 4, lr = lane & 15;
  const size_t headbase = ((size_t)(b * H_ + h)) * T_ * D_;

  {  // stage Q tile once (async: no VGPR round-trip, ASYNCcnt tracked)
    const int tx = tid & 7, ty = tid >> 3;
#pragma unroll
    for (int i = 0; i < 4; ++i) {
      const int row = ty + i * 32;
      async_ld_b128(Qs + row * 72 + tx * 8,
                    qbuf + headbase + (size_t)(q0 + row) * D_ + tx * 8);
    }
  }
  wait_async0();
  __syncthreads();

  v16bf qfrag[2];
#pragma unroll
  for (int ks = 0; ks < 2; ++ks)
    qfrag[ks] = lds_frag_a(Qs + (wave * 16 + lr) * 72 + ks * 32 + half * 8);

  v8f O[4] = {};
  float m_run[8], l_run[8];
#pragma unroll
  for (int r = 0; r < 8; ++r) { m_run[r] = -__builtin_inff(); l_run[r] = 0.f; }
  const float scale = 0.125f;  // 1/sqrt(64)

  for (int j0 = 0; j0 <= q0; j0 += 128) {
    __syncthreads();  // protect Ks/Vs from previous iteration's readers
    {  // async K tile [key][d]
      const int tx = tid & 7, ty = tid >> 3;
#pragma unroll
      for (int i = 0; i < 4; ++i) {
        const int key = ty + i * 32;
        async_ld_b128(Ks + key * 72 + tx * 8,
                      kbuf + headbase + (size_t)(j0 + key) * D_ + tx * 8);
      }
    }
    {  // V tile transposed to [d][key]
      const int tx = tid & 15, ty = tid >> 4;
#pragma unroll
      for (int i = 0; i < 8; ++i) {
        const int key = ty + i * 16;
        uint2 vv = *(const uint2*)(vbuf + headbase + (size_t)(j0 + key) * D_ + tx * 4);
        unsigned short e[4];
        *(uint2*)e = vv;
#pragma unroll
        for (int j = 0; j < 4; ++j) Vs[(tx * 4 + j) * 136 + key] = e[j];
      }
    }
    wait_async0();
    __syncthreads();

    // S = Q @ K^T : wave computes 16 rows x 128 keys = 8 accumulators
    v8f S[8];
#pragma unroll
    for (int n0 = 0; n0 < 8; ++n0) {
      v8f s = {};
#pragma unroll
      for (int ks = 0; ks < 2; ++ks) {
        v16bf kf = lds_frag_b(Ks + (n0 * 16 + lr) * 72 + ks * 32 + half * 8);
        s = WMMA_BF16(qfrag[ks], kf, s);
      }
      S[n0] = s;
    }

    // scale + causal mask on diagonal tile
    const bool diag = (j0 == q0);
#pragma unroll
    for (int n0 = 0; n0 < 8; ++n0)
#pragma unroll
      for (int r = 0; r < 8; ++r) {
        float sv = S[n0][r] * scale;
        if (diag) {
          const int key = j0 + n0 * 16 + lr;
          const int q = q0 + wave * 16 + r + half * 8;
          if (key > q) sv = -__builtin_inff();
        }
        S[n0][r] = sv;
      }

    // online softmax stats (row lives on 16 lanes of one half)
    float mnew[8], alpha[8], rsum[8];
#pragma unroll
    for (int r = 0; r < 8; ++r) {
      float v = S[0][r];
#pragma unroll
      for (int n0 = 1; n0 < 8; ++n0) v = fmaxf(v, S[n0][r]);
#pragma unroll
      for (int msk = 1; msk < 16; msk <<= 1) v = fmaxf(v, __shfl_xor(v, msk, 32));
      mnew[r] = fmaxf(m_run[r], v);
      alpha[r] = __expf(m_run[r] - mnew[r]);
      rsum[r] = 0.f;
    }

    // P = exp(S - m), store bf16 to per-wave LDS slab (A-layout reload later)
    unsigned short* Pw = Ps + wave * 16 * 136;
#pragma unroll
    for (int n0 = 0; n0 < 8; ++n0)
#pragma unroll
      for (int r = 0; r < 8; ++r) {
        const float p = __expf(S[n0][r] - mnew[r]);
        rsum[r] += p;
        Pw[(r + half * 8) * 136 + n0 * 16 + lr] = f2bf(p);
      }
#pragma unroll
    for (int r = 0; r < 8; ++r) {
      float v = rsum[r];
#pragma unroll
      for (int msk = 1; msk < 16; msk <<= 1) v += __shfl_xor(v, msk, 32);
      l_run[r] = l_run[r] * alpha[r] + v;
      m_run[r] = mnew[r];
    }
    // rescale running O
#pragma unroll
    for (int n0 = 0; n0 < 4; ++n0)
#pragma unroll
      for (int r = 0; r < 8; ++r) O[n0][r] *= alpha[r];

    __syncthreads();  // P writes visible before A-layout reload

    // O += P @ V  (K = 128 keys, 4 wmma k-steps; N = 64 head dims)
#pragma unroll
    for (int kk = 0; kk < 4; ++kk) {
      v16bf pf = lds_frag_a(Pw + lr * 136 + kk * 32 + half * 8);
#pragma unroll
      for (int n0 = 0; n0 < 4; ++n0) {
        v16bf vf = lds_frag_b(Vs + (n0 * 16 + lr) * 136 + kk * 32 + half * 8);
        O[n0] = WMMA_BF16(pf, vf, O[n0]);
      }
    }
  }

  // normalize (one rcp per row) + store bf16 attn output as [B,T,C]
#pragma unroll
  for (int r = 0; r < 8; ++r) {
    const float inv = 1.0f / l_run[r];
    const int q = q0 + wave * 16 + r + half * 8;
#pragma unroll
    for (int n0 = 0; n0 < 4; ++n0) {
      const int c = h * D_ + n0 * 16 + lr;
      attn[((size_t)(b * T_) + q) * C_ + c] = f2bf(O[n0][r] * inv);
    }
  }
}

// ---------------------------------------------------------------------------
// Kernel 3: out = attn @ w_out + b_out (fp32 output); bf16 A tile arrives
// via async global->LDS (stride 56 keeps 16B alignment + conflict-free reads)
// ---------------------------------------------------------------------------
__global__ __launch_bounds__(256) void out_gemm_kernel(
    const unsigned short* __restrict__ attn, const float* __restrict__ w,
    const float* __restrict__ bias, float* __restrict__ out) {
  __shared__ alignas(16) unsigned short As[128 * 56];  // [row][k], k-stride 56
  __shared__ alignas(16) unsigned short Bs[128 * 40];  // [n][k]
  const int n0 = blockIdx.x * 128;
  const int m0 = blockIdx.y * 128;
  const int tid = threadIdx.x;
  const int wave = tid >> 5, lane = tid & 31;
  const int half = lane >> 4, lr = lane & 15;
  const int wr = wave & 3, wc = wave >> 2;

  v8f acc[2][4] = {};

  for (int k0 = 0; k0 < C_; k0 += 32) {
    __syncthreads();
    {  // A already bf16: async 16B copies straight into LDS
      const int tx = tid & 3, ty = tid >> 2;
#pragma unroll
      for (int i = 0; i < 2; ++i) {
        const int row = ty + i * 64;
        async_ld_b128(As + row * 56 + tx * 8,
                      attn + (size_t)(m0 + row) * C_ + k0 + tx * 8);
      }
    }
    {  // B: w_out fp32 -> bf16, transposed to [n][k]
      const int nx = tid & 31, ky = tid >> 5;
#pragma unroll
      for (int i = 0; i < 4; ++i) {
        const int k = ky + i * 8;
        float4 v = *(const float4*)(w + (size_t)(k0 + k) * C_ + n0 + nx * 4);
        Bs[(nx * 4 + 0) * 40 + k] = f2bf(v.x);
        Bs[(nx * 4 + 1) * 40 + k] = f2bf(v.y);
        Bs[(nx * 4 + 2) * 40 + k] = f2bf(v.z);
        Bs[(nx * 4 + 3) * 40 + k] = f2bf(v.w);
      }
    }
    wait_async0();
    __syncthreads();
    v16bf af[2], bf[4];
#pragma unroll
    for (int mt = 0; mt < 2; ++mt)
      af[mt] = lds_frag_a(As + (wr * 32 + mt * 16 + lr) * 56 + half * 8);
#pragma unroll
    for (int nt = 0; nt < 4; ++nt)
      bf[nt] = lds_frag_b(Bs + (wc * 64 + nt * 16 + lr) * 40 + half * 8);
#pragma unroll
    for (int mt = 0; mt < 2; ++mt)
#pragma unroll
      for (int nt = 0; nt < 4; ++nt)
        acc[mt][nt] = WMMA_BF16(af[mt], bf[nt], acc[mt][nt]);
  }

#pragma unroll
  for (int mt = 0; mt < 2; ++mt)
#pragma unroll
    for (int nt = 0; nt < 4; ++nt) {
      const int n = n0 + wc * 64 + nt * 16 + lr;
      const float bval = bias[n];
#pragma unroll
      for (int r = 0; r < 8; ++r) {
        const int m = m0 + wr * 32 + mt * 16 + r + half * 8;
        out[(size_t)m * C_ + n] = acc[mt][nt][r] + bval;
      }
    }
}

// ---------------------------------------------------------------------------
extern "C" void kernel_launch(void* const* d_in, const int* in_sizes, int n_in,
                              void* d_out, int out_size, void* d_ws, size_t ws_size,
                              hipStream_t stream) {
  const float* x     = (const float*)d_in[0];
  const float* w_qkv = (const float*)d_in[1];
  const float* b_qkv = (const float*)d_in[2];
  const float* w_out = (const float*)d_in[3];
  const float* b_out = (const float*)d_in[4];
  float* out = (float*)d_out;

  // workspace: Q, K, V ([B,H,T,D] bf16) + attn ([B,T,C] bf16) = ~50 MB
  const size_t perbuf = (size_t)B_ * H_ * T_ * D_;
  unsigned short* qb   = (unsigned short*)d_ws;
  unsigned short* kb   = qb + perbuf;
  unsigned short* vb   = kb + perbuf;
  unsigned short* attn = vb + perbuf;

  qkv_gemm_kernel<<<dim3((3 * C_) / 128, (B_ * T_) / 128), 256, 0, stream>>>(
      x, w_qkv, b_qkv, qb, kb, vb);
  attn_kernel<<<dim3((T_ / 128) * B_ * H_), 256, 0, stream>>>(qb, kb, vb, attn);
  out_gemm_kernel<<<dim3(C_ / 128, (B_ * T_) / 128), 256, 0, stream>>>(
      attn, w_out, b_out, out);
}
